// Predictor_block_24644522344596
// MI455X (gfx1250) — matrix operand order repaired and checked by reading, not compiled, not run on hardware
//
#include <hip/hip_runtime.h>

// ---------------------------------------------------------------------------
// LSTM encoder + autoregressive rollout for MI455X (gfx1250, wave32, WMMA).
//   B=2048, S=64, H=256, 4H=1024, n_steps derived from out_size.
// Per step:  gates[2048,1024] = A0 @ Wa^T (+ A1 @ Wb^T) + bias, then fused
// LSTM cell.  bf16 WMMA 16x16x32, fp32 accumulate, fp32 cell state.
// Weights for the block's N-slice are staged into LDS (async copy), padded
// to kill ds bank conflicts; A operands stream from global (L2-resident).
// ---------------------------------------------------------------------------

#define USE_ASYNC_COPY 1

typedef __attribute__((ext_vector_type(16))) __bf16 v16bf;
typedef __attribute__((ext_vector_type(8)))  float  v8f;

union FragU { uint4 u[2]; v16bf v; };

// padded LDS row stride (shorts): 256 data + 8 pad -> 528B = 132 DWORDs
constexpr int LROW = 264;

__device__ __forceinline__ unsigned short f2bf(float f) {
    union { float f; unsigned u; } x; x.f = f;
    unsigned r = x.u + 0x7FFFu + ((x.u >> 16) & 1u);   // round-to-nearest-even
    return (unsigned short)(r >> 16);
}

__device__ __forceinline__ float fast_sigmoid(float x) {
    return 1.0f / (1.0f + __expf(-x));
}
__device__ __forceinline__ float fast_tanh(float x) {
    float ax = fabsf(x);
    float e  = __expf(-2.0f * ax);
    float t  = (1.0f - e) / (1.0f + e);
    return copysignf(t, x);
}

// A-matrix fragment (16x32 bf16, ISA 7.12.2): lane L holds row (L&15);
// K chunks [ (L>>4)*8 , +8 ) and [ 16+(L>>4)*8 , +8 )  -> two b128 loads.
__device__ __forceinline__ v16bf load_a_frag(const unsigned short* __restrict__ A,
                                             int lda, int row0, int lane, int kc) {
    const unsigned short* p =
        A + (size_t)(row0 + (lane & 15)) * lda + kc + ((lane >> 4) << 3);
    FragU f;
    f.u[0] = *(const uint4*)(p);        // K base .. base+7
    f.u[1] = *(const uint4*)(p + 16);   // K base+16 .. base+23
    return f.v;
}

// B-matrix fragment (32x16 bf16) from padded LDS: lane L holds column (L&15);
// K chunks [ (L>>4)*8 , +8 ) and [ 16+(L>>4)*8 , +8 ).
__device__ __forceinline__ v16bf lds_b_frag(const unsigned short* sW,
                                            int lrow, int lane, int kc) {
    const unsigned short* p =
        sW + (lrow + (lane & 15)) * LROW + kc + ((lane >> 4) << 3);
    FragU f;
    f.u[0] = *(const uint4*)(p);        // K base .. base+7
    f.u[1] = *(const uint4*)(p + 16);   // K base+16 .. base+23
    return f.v;
}

// 16-byte global -> LDS copy (per lane) via the CDNA5 async engine.
__device__ __forceinline__ void async_copy16(unsigned short* ldsp,
                                             const unsigned short* gp) {
#if USE_ASYNC_COPY
    asm volatile("global_load_async_to_lds_b128 %0, %1, off"
                 :: "v"((unsigned)(uintptr_t)ldsp), "v"(gp)
                 : "memory");
#else
    *(uint4*)ldsp = *(const uint4*)gp;
#endif
}

__device__ __forceinline__ void async_copy_fence() {
#if USE_ASYNC_COPY
    asm volatile("s_wait_asynccnt 0x0" ::: "memory");
#endif
}

// --------------------------- one-time prep kernels -------------------------

__global__ __launch_bounds__(256)
void prep_weights(const float* __restrict__ wih, const float* __restrict__ whh,
                  const float* __restrict__ bih, const float* __restrict__ bhh,
                  unsigned short* __restrict__ wih_b, unsigned short* __restrict__ whh_b,
                  unsigned short* __restrict__ wsum_b, float* __restrict__ bias) {
    int i = blockIdx.x * 256 + threadIdx.x;     // 1024*256 elements
    float a = wih[i], b = whh[i];
    wih_b[i]  = f2bf(a);
    whh_b[i]  = f2bf(b);
    wsum_b[i] = f2bf(a + b);
    if (i < 1024) bias[i] = bih[i] + bhh[i];
}

__global__ __launch_bounds__(256)
void convert_x(const float4* __restrict__ in, uint2* __restrict__ out) {
    int i = blockIdx.x * 256 + threadIdx.x;     // n/4 float4 elements
    float4 v = in[i];
    uint2 o;
    o.x = (unsigned)f2bf(v.x) | ((unsigned)f2bf(v.y) << 16);
    o.y = (unsigned)f2bf(v.z) | ((unsigned)f2bf(v.w) << 16);
    out[i] = o;
}

__global__ __launch_bounds__(256)
void zero_state(float* __restrict__ c, unsigned short* __restrict__ h) {
    int i = blockIdx.x * 256 + threadIdx.x;     // 2048*256 elements
    c[i] = 0.0f;
    h[i] = 0;
}

// ------------------------------- step kernel -------------------------------
// grid = (8, 16): blockIdx.x -> 32 per-gate columns, blockIdx.y -> 128 rows.
// 256 threads = 8 waves; wave w owns rows [m0+16w, m0+16w+16).
// Each wave accumulates 4 gates x 2 col-subtiles = 8 WMMA tiles (64 VGPRs).
// The block's 128 weight rows (4 gates x 32 cols) per source live in LDS.

template<bool DUAL>
__global__ __launch_bounds__(256)
void lstm_step(const unsigned short* __restrict__ A0, int lda0,   // x_t (DUAL only)
               const unsigned short* __restrict__ Hprev,          // [2048,256] bf16
               const unsigned short* __restrict__ Wa,             // W_ih or W_sum
               const unsigned short* __restrict__ Wb,             // W_hh (DUAL only)
               const float* __restrict__ bias,                    // [1024] fused
               float* __restrict__ C,                             // cell [2048,256]
               unsigned short* __restrict__ Hnext,                // [2048,256] bf16
               float* __restrict__ Out, int outStride)            // nullable
{
    __shared__ unsigned short sW[(DUAL ? 2 : 1) * 128 * LROW];

    const int tid  = threadIdx.x;
    const int lane = tid & 31;
    const int wave = tid >> 5;
    const int m0   = blockIdx.y * 128 + wave * 16;   // row base for this wave
    const int nb   = blockIdx.x * 32;                // per-gate column base

    // ---- stage weight slice(s) into LDS: 128 rows x 256 K, 16B chunks ----
    #pragma unroll
    for (int i = tid; i < 128 * 32; i += 256) {      // 32 chunks of 8 shorts/row
        const int row = i >> 5;                      // 0..127 (g*32 + r)
        const int ch  = (i & 31) << 3;               // short offset in row
        const int g   = row >> 5;
        const int r   = row & 31;
        const size_t goff = (size_t)(g * 256 + nb + r) * 256 + ch;
        async_copy16(&sW[row * LROW + ch], Wa + goff);
        if constexpr (DUAL)
            async_copy16(&sW[128 * LROW + row * LROW + ch], Wb + goff);
    }
    async_copy_fence();
    __syncthreads();

    v8f acc[4][2] = {};

    for (int kc = 0; kc < 256; kc += 32) {
        if constexpr (DUAL) {
            v16bf a0 = load_a_frag(A0, lda0, m0, lane, kc);
            #pragma unroll
            for (int g = 0; g < 4; ++g)
                #pragma unroll
                for (int ns = 0; ns < 2; ++ns) {
                    v16bf b = lds_b_frag(sW, g * 32 + ns * 16, lane, kc);
                    acc[g][ns] = __builtin_amdgcn_wmma_f32_16x16x32_bf16(
                        false, a0, false, b, (short)0, acc[g][ns], false, false);
                }
        }
        {
            v16bf a1 = load_a_frag(Hprev, 256, m0, lane, kc);
            const unsigned short* sW2 = DUAL ? sW + 128 * LROW : sW;
            #pragma unroll
            for (int g = 0; g < 4; ++g)
                #pragma unroll
                for (int ns = 0; ns < 2; ++ns) {
                    v16bf b = lds_b_frag(sW2, g * 32 + ns * 16, lane, kc);
                    acc[g][ns] = __builtin_amdgcn_wmma_f32_16x16x32_bf16(
                        false, a1, false, b, (short)0, acc[g][ns], false, false);
                }
        }
    }

    // Fused LSTM cell.  C/D layout (ISA 7.12.2): lane L holds col (L&15);
    // VGPR r holds row r (+8 for lanes 16..31).
    const int mrow = m0 + ((lane >> 4) << 3);
    #pragma unroll
    for (int ns = 0; ns < 2; ++ns) {
        const int col = nb + ns * 16 + (lane & 15);      // 0..255 within gate
        const float bi = bias[col];
        const float bf = bias[256 + col];
        const float bg = bias[512 + col];
        const float bo = bias[768 + col];
        #pragma unroll
        for (int r = 0; r < 8; ++r) {
            const int m = mrow + r;
            const float iv = fast_sigmoid(acc[0][ns][r] + bi);
            const float fv = fast_sigmoid(acc[1][ns][r] + bf);
            const float gv = fast_tanh   (acc[2][ns][r] + bg);
            const float ov = fast_sigmoid(acc[3][ns][r] + bo);
            const size_t idx = (size_t)m * 256 + col;
            const float cn = fv * C[idx] + iv * gv;
            C[idx] = cn;
            const float hn = ov * fast_tanh(cn);
            Hnext[idx] = f2bf(hn);
            if (Out) Out[(size_t)m * outStride + col] = hn;
        }
    }
}

// ------------------------------- host driver -------------------------------

extern "C" void kernel_launch(void* const* d_in, const int* in_sizes, int n_in,
                              void* d_out, int out_size, void* d_ws, size_t ws_size,
                              hipStream_t stream) {
    constexpr int B = 2048, S = 64, H = 256, G4H = 1024;
    const float* x0   = (const float*)d_in[0];
    const float* W_ih = (const float*)d_in[1];
    const float* W_hh = (const float*)d_in[2];
    const float* b_ih = (const float*)d_in[3];
    const float* b_hh = (const float*)d_in[4];
    float* out = (float*)d_out;

    const int n_steps   = out_size / (B * H) - 1;   // reference: 32
    const int outStride = (n_steps + 1) * H;

    // Workspace carve-up (~70 MB total).
    char* w = (char*)d_ws;
    unsigned short* wih_b  = (unsigned short*)(w);                       // 512 KB
    unsigned short* whh_b  = (unsigned short*)(w + (512u << 10));        // 512 KB
    unsigned short* wsum_b = (unsigned short*)(w + (1024u << 10));       // 512 KB
    float*          bias   = (float*)(w + (1536u << 10));                //   4 KB
    unsigned short* x_b    = (unsigned short*)(w + (1540u << 10));       //  64 MB
    size_t off = (1540u << 10) + (size_t)B * S * H * 2;
    unsigned short* h_ping = (unsigned short*)(w + off);                 //   1 MB
    unsigned short* h_pong = (unsigned short*)(w + off + (size_t)B * H * 2);
    float*          cstate = (float*)(w + off + (size_t)B * H * 4);      //   2 MB

    // One-time preparation (all on stream; graph-capture safe).
    prep_weights<<<(G4H * H) / 256, 256, 0, stream>>>(W_ih, W_hh, b_ih, b_hh,
                                                      wih_b, whh_b, wsum_b, bias);
    convert_x<<<((size_t)B * S * H / 4) / 256, 256, 0, stream>>>(
        (const float4*)x0, (uint2*)x_b);
    zero_state<<<(B * H) / 256, 256, 0, stream>>>(cstate, h_ping);

    dim3 grid(H / 32, B / 128);   // (8, 16)
    unsigned short* hp = h_ping;
    unsigned short* hn = h_pong;

    // Encoder: 64 steps; last step's h is prediction[:, 0, :].
    for (int t = 0; t < S; ++t) {
        float* o = (t == S - 1) ? out : nullptr;
        lstm_step<true><<<grid, 256, 0, stream>>>(
            x_b + (size_t)t * H, S * H, hp, wih_b, whh_b, bias, cstate, hn, o, outStride);
        unsigned short* tmp = hp; hp = hn; hn = tmp;
    }
    // Autoregressive rollout: gates = h @ (W_ih + W_hh)^T.
    for (int t = 0; t < n_steps; ++t) {
        lstm_step<false><<<grid, 256, 0, stream>>>(
            nullptr, 0, hp, wsum_b, nullptr, bias, cstate, hn,
            out + (size_t)(t + 1) * H, outStride);
        unsigned short* tmp = hp; hp = hn; hn = tmp;
    }
}
